// SingleDecoder_20143396618854
// MI455X (gfx1250) — matrix operand order repaired and checked by reading, not compiled, not run on hardware
//
#include <hip/hip_runtime.h>

typedef __attribute__((ext_vector_type(16))) __bf16 v16bf;
typedef __attribute__((ext_vector_type(8)))  float  v8f;

#define L_SEQ    600
#define BATCH    64
#define DSPEC    552
#define DSPEC_P  576
#define DPRE     256
#define DH       1024
#define G4       4096
#define LIN_NP   560
#define CMID     512
#define LPAD     624
#define LT_TILES 38
#define NB_LSTM  32
#define MROWS    (L_SEQ * BATCH)   /* 38400 */
#define MTILES   (MROWS / 16)      /* 2400  */

struct alignas(16) U4 { unsigned x, y, z, w; };
union FragU { v16bf v; U4 q[2]; };

__device__ __forceinline__ __bf16 f2bf(float f) {
  unsigned u = __builtin_bit_cast(unsigned, f);
  u += 0x7FFFu + ((u >> 16) & 1u);
  unsigned short s = (unsigned short)(u >> 16);
  return __builtin_bit_cast(__bf16, s);
}

// A fragment (16x32 bf16, MxK): lane m = lane&15, k-base = 8*(lane>>4);
// elems 0..7 = k..k+7, elems 8..15 = k+16..k+23   (ISA 7.12.2)
__device__ __forceinline__ v16bf frag_a(const __bf16* p0, int lda, int lane) {
  const __bf16* p = p0 + (size_t)(lane & 15) * lda + ((lane >> 4) << 3);
  FragU u;
  u.q[0] = *(const U4*)(p);
  u.q[1] = *(const U4*)(p + 16);
  return u.v;
}
// B fragment (32x16 bf16, KxN) from W[N][K] row-major: lane n = lane&15,
// k-base = 8*(lane>>4); elems 0..7 = k..k+7, elems 8..15 = k+16..k+23.
__device__ __forceinline__ v16bf frag_b(const __bf16* __restrict__ p0, int ldb, int lane) {
  const __bf16* p = p0 + (size_t)(lane & 15) * ldb + ((lane >> 4) << 3);
  FragU u;
  u.q[0] = *(const U4*)(p);
  u.q[1] = *(const U4*)(p + 16);
  return u.v;
}

#define WMMA_BF16(a, b, c) \
  __builtin_amdgcn_wmma_f32_16x16x32_bf16(false, (a), false, (b), (short)0, (c), false, false)

__device__ __forceinline__ float sigm(float x) { return 1.0f / (1.0f + __expf(-x)); }

// ---------------------------------------------------------------- prep kernels

__global__ void zero_kernel(U4* __restrict__ p, long n) {
  long step = (long)gridDim.x * blockDim.x;
  for (long i = (long)blockIdx.x * blockDim.x + threadIdx.x; i < n; i += step) {
    U4 z; z.x = 0u; z.y = 0u; z.z = 0u; z.w = 0u;
    p[i] = z;
  }
}

// dst[dr][dp] bf16 <- src[sr][sp] f32 copying cc cols, zero-padding elsewhere
__global__ void cvt_pad_kernel(__bf16* __restrict__ dst, const float* __restrict__ src,
                               int dr, int dp, int sr, int sp, int cc) {
  long n = (long)dr * dp;
  long step = (long)gridDim.x * blockDim.x;
  for (long i = (long)blockIdx.x * blockDim.x + threadIdx.x; i < n; i += step) {
    int r = (int)(i / dp), c = (int)(i - (long)r * dp);
    float v = (r < sr && c < cc) ? src[(size_t)r * sp + c] : 0.0f;
    dst[i] = f2bf(v);
  }
}

// dst[t][coutp][cinp] bf16 <- src[cout][cin][5] f32
__global__ void conv_pack_kernel(__bf16* __restrict__ dst, const float* __restrict__ src,
                                 int coutp, int cout, int cinp, int cin) {
  long n = 5L * coutp * cinp;
  long step = (long)gridDim.x * blockDim.x;
  long plane = (long)coutp * cinp;
  for (long i = (long)blockIdx.x * blockDim.x + threadIdx.x; i < n; i += step) {
    int t = (int)(i / plane);
    long rem = i - (long)t * plane;
    int co = (int)(rem / cinp), ci = (int)(rem - (long)co * cinp);
    float v = (co < cout && ci < cin) ? src[((size_t)co * cin + ci) * 5 + t] : 0.0f;
    dst[i] = f2bf(v);
  }
}

__global__ void bias_sum_kernel(float* __restrict__ d, const float* __restrict__ a,
                                const float* __restrict__ b, int n) {
  int step = gridDim.x * blockDim.x;
  for (int i = blockIdx.x * blockDim.x + threadIdx.x; i < n; i += step) d[i] = a[i] + b[i];
}

// in_frames[t] = S_pad[t-1] (zeros at t=0), padded 552->576, bf16
__global__ void build_inbf_kernel(const float* __restrict__ S, __bf16* __restrict__ inbf) {
  long n = (long)MROWS * DSPEC_P;
  long step = (long)gridDim.x * blockDim.x;
  for (long i = (long)blockIdx.x * blockDim.x + threadIdx.x; i < n; i += step) {
    int row = (int)(i / DSPEC_P), c = (int)(i - (long)row * DSPEC_P);
    int t = row >> 6, b = row & 63;
    float v = (t > 0 && c < DSPEC) ? S[((size_t)(t - 1) * BATCH + b) * DSPEC + c] : 0.0f;
    inbf[i] = f2bf(v);
  }
}

// ---------------------------------------------------------------- prenet GEMMs
__global__ void gemm_relu_kernel(const __bf16* __restrict__ A, int K,
                                 const __bf16* __restrict__ W,
                                 const float* __restrict__ bias,
                                 __bf16* __restrict__ out) {
  int lane = threadIdx.x & 31;
  int wave = blockIdx.x * (blockDim.x >> 5) + (threadIdx.x >> 5);
  const int ntiles = DPRE / 16;
  if (wave >= MTILES * ntiles) return;
  int mt = wave % MTILES, nt = wave / MTILES;
  v8f acc = {};
  const __bf16* Ab = A + (size_t)(mt * 16) * K;
  const __bf16* Bw = W + (size_t)(nt * 16) * K;
  for (int kb = 0; kb < K / 32; ++kb) {
    v16bf a = frag_a(Ab + kb * 32, K, lane);
    v16bf b = frag_b(Bw + kb * 32, K, lane);
    acc = WMMA_BF16(a, b, acc);
  }
  int n = nt * 16 + (lane & 15);
  float bv = bias[n];
#pragma unroll
  for (int r = 0; r < 8; ++r) {
    int m = mt * 16 + r + ((lane >> 4) << 3);
    float v = acc[r] + bv;
    v = v > 0.0f ? v : 0.0f;
    out[(size_t)m * DPRE + n] = f2bf(v);
  }
}

// ---------------------------------------------------------------- LSTM (persistent)

__device__ __forceinline__ void gbar(unsigned* bar) {
  __syncthreads();
  if (threadIdx.x == 0) {
    __threadfence();
    unsigned gen = atomicAdd(&bar[1], 0u);
    if (atomicAdd(&bar[0], 1u) == gridDim.x - 1) {
      atomicExch(&bar[0], 0u);
      atomicAdd(&bar[1], 1u);
    } else {
      while (atomicAdd(&bar[1], 0u) == gen) { __builtin_amdgcn_s_sleep(2); }
    }
    __threadfence();
  }
  __syncthreads();
}

// Block b: mt = b&3 (batch tile), ntg = b>>2; wave w handles nt = ntg*8+w
// (channel range nt*16..nt*16+15). Each wave computes all 4 gate quadrants
// i/f/g/o for its (mt,nt) so the LSTM cell runs in-register (no gates buffer).
// A slab (16 x K) is staged once per block into LDS and shared by all 8 waves.
__global__ void __launch_bounds__(256, 1)
lstm_kernel(const __bf16* __restrict__ prebf,
            const __bf16* __restrict__ Wih0, const __bf16* __restrict__ Whh0,
            const float* __restrict__ bias0,
            const __bf16* __restrict__ Wih1, const __bf16* __restrict__ Whh1,
            const float* __restrict__ bias1,
            __bf16* __restrict__ h0buf, float* __restrict__ c0,
            __bf16* __restrict__ h1buf, float* __restrict__ c1,
            __bf16* __restrict__ outsbf,
            unsigned* __restrict__ bar) {
  __shared__ U4 sAu[16 * 256];                 // 64 KB: 16 rows x 2048 bf16 max
  const __bf16* sAbf = (const __bf16*)sAu;

  const int lane = threadIdx.x & 31;
  const int wv = threadIdx.x >> 5;             // 0..7
  const int mt = blockIdx.x & 3;
  const int ntg = blockIdx.x >> 2;             // 0..7
  const int nt = ntg * 8 + wv;                 // 0..63
  const int tid = threadIdx.x;

  const int c = nt * 16 + (lane & 15);         // hidden channel of this lane
  const float b0i = bias0[c], b0f = bias0[c + 1024], b0g = bias0[c + 2048], b0o = bias0[c + 3072];
  const float b1i = bias1[c], b1f = bias1[c + 1024], b1g = bias1[c + 2048], b1o = bias1[c + 3072];

  // quadrant weight row bases
  const __bf16* Wi0q[4]; const __bf16* Wh0q[4];
  const __bf16* Wi1q[4]; const __bf16* Wh1q[4];
#pragma unroll
  for (int q = 0; q < 4; ++q) {
    Wi0q[q] = Wih0 + (size_t)(q * 1024 + nt * 16) * DPRE;
    Wh0q[q] = Whh0 + (size_t)(q * 1024 + nt * 16) * DH;
    Wi1q[q] = Wih1 + (size_t)(q * 1024 + nt * 16) * DH;
    Wh1q[q] = Whh1 + (size_t)(q * 1024 + nt * 16) * DH;
  }

  for (int t = 0; t < L_SEQ; ++t) {
    const __bf16* h0p = h0buf + (size_t)(t & 1) * (BATCH * DH);
    __bf16*       h0c = h0buf + (size_t)((t + 1) & 1) * (BATCH * DH);
    const __bf16* h1p = h1buf + (size_t)(t & 1) * (BATCH * DH);
    __bf16*       h1c = h1buf + (size_t)((t + 1) & 1) * (BATCH * DH);

    // ---- stage layer0 A slab: rows mt*16.., cols [pre_t(256) | h0p(1024)]
    {
      const __bf16* xsrc = prebf + (size_t)(t * BATCH + mt * 16) * DPRE;
      for (int j = tid; j < 16 * 160; j += 256) {      // 160 U4 per 1280-col row
        int row = j / 160, col = j - row * 160;
        U4 v;
        if (col < 32) v = ((const U4*)(xsrc + (size_t)row * DPRE))[col];
        else          v = ((const U4*)(h0p + (size_t)(mt * 16 + row) * DH))[col - 32];
        sAu[row * 160 + col] = v;
      }
    }
    __syncthreads();

    // ---- layer0: 4 quadrant tiles, K = 1280
    {
      v8f ai = {}, af = {}, ag = {}, ao = {};
      for (int kb = 0; kb < 40; ++kb) {
        v16bf a = frag_a(sAbf + kb * 32, 1280, lane);
        const __bf16 *p0, *p1, *p2, *p3; int ld;
        if (kb < 8) { int o = kb * 32; ld = DPRE;
          p0 = Wi0q[0] + o; p1 = Wi0q[1] + o; p2 = Wi0q[2] + o; p3 = Wi0q[3] + o;
        } else { int o = (kb - 8) * 32; ld = DH;
          p0 = Wh0q[0] + o; p1 = Wh0q[1] + o; p2 = Wh0q[2] + o; p3 = Wh0q[3] + o;
        }
        ai = WMMA_BF16(a, frag_b(p0, ld, lane), ai);
        af = WMMA_BF16(a, frag_b(p1, ld, lane), af);
        ag = WMMA_BF16(a, frag_b(p2, ld, lane), ag);
        ao = WMMA_BF16(a, frag_b(p3, ld, lane), ao);
      }
      // in-register cell0
#pragma unroll
      for (int r = 0; r < 8; ++r) {
        int bb = mt * 16 + r + ((lane >> 4) << 3);
        size_t idx = (size_t)bb * DH + c;
        float cn = sigm(af[r] + b0f) * c0[idx] + sigm(ai[r] + b0i) * tanhf(ag[r] + b0g);
        float h  = sigm(ao[r] + b0o) * tanhf(cn);
        c0[idx] = cn;
        h0c[idx] = f2bf(h);
      }
    }
    gbar(bar);

    // ---- stage layer1 A slab: rows mt*16.., cols [h0c(1024) | h1p(1024)]
    for (int j = tid; j < 16 * 256; j += 256) {        // 256 U4 per 2048-col row
      int row = j >> 8, col = j & 255;
      U4 v;
      if (col < 128) v = ((const U4*)(h0c + (size_t)(mt * 16 + row) * DH))[col];
      else           v = ((const U4*)(h1p + (size_t)(mt * 16 + row) * DH))[col - 128];
      sAu[row * 256 + col] = v;
    }
    __syncthreads();

    // ---- layer1: 4 quadrant tiles, K = 2048
    {
      v8f ai = {}, af = {}, ag = {}, ao = {};
      for (int kb = 0; kb < 64; ++kb) {
        v16bf a = frag_a(sAbf + kb * 32, 2048, lane);
        const __bf16 *p0, *p1, *p2, *p3;
        if (kb < 32) { int o = kb * 32;
          p0 = Wi1q[0] + o; p1 = Wi1q[1] + o; p2 = Wi1q[2] + o; p3 = Wi1q[3] + o;
        } else { int o = (kb - 32) * 32;
          p0 = Wh1q[0] + o; p1 = Wh1q[1] + o; p2 = Wh1q[2] + o; p3 = Wh1q[3] + o;
        }
        ai = WMMA_BF16(a, frag_b(p0, DH, lane), ai);
        af = WMMA_BF16(a, frag_b(p1, DH, lane), af);
        ag = WMMA_BF16(a, frag_b(p2, DH, lane), ag);
        ao = WMMA_BF16(a, frag_b(p3, DH, lane), ao);
      }
      // in-register cell1 (+ emit h1 to outs)
#pragma unroll
      for (int r = 0; r < 8; ++r) {
        int bb = mt * 16 + r + ((lane >> 4) << 3);
        size_t idx = (size_t)bb * DH + c;
        float cn = sigm(af[r] + b1f) * c1[idx] + sigm(ai[r] + b1i) * tanhf(ag[r] + b1g);
        float h  = sigm(ao[r] + b1o) * tanhf(cn);
        c1[idx] = cn;
        __bf16 hb = f2bf(h);
        h1c[idx] = hb;
        outsbf[(size_t)t * (BATCH * DH) + idx] = hb;
      }
    }
    gbar(bar);
  }
}

// ---------------------------------------------------------------- projection
__global__ void linear_kernel(const __bf16* __restrict__ outsbf,
                              const __bf16* __restrict__ linW,
                              const float* __restrict__ lin_b,
                              float* __restrict__ dout,
                              __bf16* __restrict__ xpadA) {
  int lane = threadIdx.x & 31;
  int wave = blockIdx.x * (blockDim.x >> 5) + (threadIdx.x >> 5);
  const int ntiles = LIN_NP / 16;   // 35
  if (wave >= MTILES * ntiles) return;
  int mt = wave % MTILES, nt = wave / MTILES;
  v8f acc = {};
  const __bf16* A  = outsbf + (size_t)(mt * 16) * DH;
  const __bf16* Bw = linW + (size_t)(nt * 16) * DH;
  for (int kb = 0; kb < DH / 32; ++kb) {
    v16bf a = frag_a(A + kb * 32, DH, lane);
    v16bf b = frag_b(Bw + kb * 32, DH, lane);
    acc = WMMA_BF16(a, b, acc);
  }
  int n = nt * 16 + (lane & 15);
  float bv = (n < DSPEC) ? lin_b[n] : 0.0f;
#pragma unroll
  for (int r = 0; r < 8; ++r) {
    int row = mt * 16 + r + ((lane >> 4) << 3);
    float v = acc[r] + bv;
    if (n < DSPEC) {
      dout[(size_t)row * DSPEC + n] = v;
      int t = row >> 6, b = row & 63;
      xpadA[((size_t)b * LPAD + (t + 2)) * DSPEC_P + n] = f2bf(v);
    }
  }
}

// ---------------------------------------------------------------- postnet conv
__global__ void conv_kernel(const __bf16* __restrict__ xin, int cinp,
                            const __bf16* __restrict__ wp, int coutp, int cout,
                            const float* __restrict__ bias,
                            __bf16* __restrict__ xout,
                            float* __restrict__ dout, int do_tanh) {
  int lane = threadIdx.x & 31;
  int wave = blockIdx.x * (blockDim.x >> 5) + (threadIdx.x >> 5);
  int ntiles = coutp >> 4;
  int total = BATCH * LT_TILES * ntiles;
  if (wave >= total) return;
  int nt = wave % ntiles;
  int mtile = wave / ntiles;
  int b = mtile / LT_TILES;
  int l0 = (mtile % LT_TILES) * 16;
  v8f acc = {};
  const __bf16* Abase = xin + ((size_t)b * LPAD + l0) * cinp;
  for (int ts = 0; ts < 5; ++ts) {
    const __bf16* At = Abase + (size_t)ts * cinp;
    const __bf16* Bt = wp + ((size_t)ts * coutp + nt * 16) * cinp;
    for (int kb = 0; kb < cinp / 32; ++kb) {
      v16bf a = frag_a(At + kb * 32, cinp, lane);
      v16bf bb = frag_b(Bt + kb * 32, cinp, lane);
      acc = WMMA_BF16(a, bb, acc);
    }
  }
  int n = nt * 16 + (lane & 15);
  float bv = (n < cout) ? bias[n] : 0.0f;
#pragma unroll
  for (int r = 0; r < 8; ++r) {
    int l = l0 + r + ((lane >> 4) << 3);
    float v = acc[r] + bv;
    if (l < L_SEQ && n < cout) {
      if (dout) {
        dout[((size_t)l * BATCH + b) * DSPEC + n] += v;   // residual add
      } else {
        if (do_tanh) v = tanhf(v);
        xout[((size_t)b * LPAD + (l + 2)) * coutp + n] = f2bf(v);
      }
    }
  }
}

// ---------------------------------------------------------------- launcher

extern "C" void kernel_launch(void* const* d_in, const int* in_sizes, int n_in,
                              void* d_out, int out_size, void* d_ws, size_t ws_size,
                              hipStream_t stream) {
  (void)in_sizes; (void)n_in; (void)out_size; (void)ws_size;
  const float* S_pad = (const float*)d_in[0];
  const float* pre_W1 = (const float*)d_in[1];
  const float* pre_b1 = (const float*)d_in[2];
  const float* pre_W2 = (const float*)d_in[3];
  const float* pre_b2 = (const float*)d_in[4];
  const float* Wih0 = (const float*)d_in[5];
  const float* Whh0 = (const float*)d_in[6];
  const float* bih0 = (const float*)d_in[7];
  const float* bhh0 = (const float*)d_in[8];
  const float* Wih1 = (const float*)d_in[9];
  const float* Whh1 = (const float*)d_in[10];
  const float* bih1 = (const float*)d_in[11];
  const float* bhh1 = (const float*)d_in[12];
  const float* lin_W = (const float*)d_in[13];
  const float* lin_b = (const float*)d_in[14];
  const float* cw1 = (const float*)d_in[15];
  const float* cb1 = (const float*)d_in[16];
  const float* cw2 = (const float*)d_in[17];
  const float* cb2 = (const float*)d_in[18];
  const float* cw3 = (const float*)d_in[19];
  const float* cb3 = (const float*)d_in[20];
  const float* cw4 = (const float*)d_in[21];
  const float* cb4 = (const float*)d_in[22];
  const float* cw5 = (const float*)d_in[23];
  const float* cb5 = (const float*)d_in[24];
  float* dout = (float*)d_out;

  char* ws = (char*)d_ws;
  size_t off = 0;
  auto alloc = [&](size_t bytes) -> char* {
    char* p = ws + off;
    off += (bytes + 255) & ~(size_t)255;
    return p;
  };

  unsigned* bar   = (unsigned*)alloc(64);
  __bf16* inbf    = (__bf16*)alloc((size_t)MROWS * DSPEC_P * 2);
  __bf16* preW1bf = (__bf16*)alloc((size_t)DPRE * DSPEC_P * 2);
  __bf16* preW2bf = (__bf16*)alloc((size_t)DPRE * DPRE * 2);
  __bf16* pre1bf  = (__bf16*)alloc((size_t)MROWS * DPRE * 2);
  __bf16* prebf   = (__bf16*)alloc((size_t)MROWS * DPRE * 2);
  __bf16* Wih0bf  = (__bf16*)alloc((size_t)G4 * DPRE * 2);
  __bf16* Whh0bf  = (__bf16*)alloc((size_t)G4 * DH * 2);
  __bf16* Wih1bf  = (__bf16*)alloc((size_t)G4 * DH * 2);
  __bf16* Whh1bf  = (__bf16*)alloc((size_t)G4 * DH * 2);
  float* bias0    = (float*)alloc((size_t)G4 * 4);
  float* bias1    = (float*)alloc((size_t)G4 * 4);
  __bf16* h0buf   = (__bf16*)alloc(2 * (size_t)BATCH * DH * 2);   // double-buffered
  __bf16* h1buf   = (__bf16*)alloc(2 * (size_t)BATCH * DH * 2);
  float* c0       = (float*)alloc((size_t)BATCH * DH * 4);
  float* c1       = (float*)alloc((size_t)BATCH * DH * 4);
  __bf16* outsbf  = (__bf16*)alloc((size_t)L_SEQ * BATCH * DH * 2);
  __bf16* linWbf  = (__bf16*)alloc((size_t)LIN_NP * DH * 2);
  __bf16* xpadA   = (__bf16*)alloc((size_t)BATCH * LPAD * DSPEC_P * 2);
  __bf16* xpadB   = (__bf16*)alloc((size_t)BATCH * LPAD * CMID * 2);
  __bf16* xpadC   = (__bf16*)alloc((size_t)BATCH * LPAD * CMID * 2);
  __bf16* cw1p    = (__bf16*)alloc(5 * (size_t)CMID * DSPEC_P * 2);
  __bf16* cw2p    = (__bf16*)alloc(5 * (size_t)CMID * CMID * 2);
  __bf16* cw3p    = (__bf16*)alloc(5 * (size_t)CMID * CMID * 2);
  __bf16* cw4p    = (__bf16*)alloc(5 * (size_t)CMID * CMID * 2);
  __bf16* cw5p    = (__bf16*)alloc(5 * (size_t)LIN_NP * CMID * 2);

  auto Z = [&](void* p, size_t bytes) {
    long n = (long)(bytes / 16);
    int blocks = (int)((n + 255) / 256); if (blocks > 4096) blocks = 4096; if (blocks < 1) blocks = 1;
    zero_kernel<<<blocks, 256, 0, stream>>>((U4*)p, n);
  };
  auto EW = [&](long n) {
    int blocks = (int)((n + 255) / 256); if (blocks > 4096) blocks = 4096; if (blocks < 1) blocks = 1;
    return blocks;
  };

  // state / margins that must start at zero every call
  Z(bar, 64);
  Z(h0buf, 2 * (size_t)BATCH * DH * 2);
  Z(h1buf, 2 * (size_t)BATCH * DH * 2);
  Z(c0, (size_t)BATCH * DH * 4);
  Z(c1, (size_t)BATCH * DH * 4);
  Z(xpadA, (size_t)BATCH * LPAD * DSPEC_P * 2);
  Z(xpadB, (size_t)BATCH * LPAD * CMID * 2);
  Z(xpadC, (size_t)BATCH * LPAD * CMID * 2);

  // weight conversion / packing (bf16, padded, [N][K] layouts)
  cvt_pad_kernel<<<EW((long)DPRE * DSPEC_P), 256, 0, stream>>>(preW1bf, pre_W1, DPRE, DSPEC_P, DPRE, DSPEC, DSPEC);
  cvt_pad_kernel<<<EW((long)DPRE * DPRE), 256, 0, stream>>>(preW2bf, pre_W2, DPRE, DPRE, DPRE, DPRE, DPRE);
  cvt_pad_kernel<<<EW((long)G4 * DPRE), 256, 0, stream>>>(Wih0bf, Wih0, G4, DPRE, G4, DPRE + 128, DPRE);  // drop zero-ctx cols
  cvt_pad_kernel<<<EW((long)G4 * DH), 256, 0, stream>>>(Whh0bf, Whh0, G4, DH, G4, DH, DH);
  cvt_pad_kernel<<<EW((long)G4 * DH), 256, 0, stream>>>(Wih1bf, Wih1, G4, DH, G4, DH, DH);
  cvt_pad_kernel<<<EW((long)G4 * DH), 256, 0, stream>>>(Whh1bf, Whh1, G4, DH, G4, DH, DH);
  cvt_pad_kernel<<<EW((long)LIN_NP * DH), 256, 0, stream>>>(linWbf, lin_W, LIN_NP, DH, DSPEC, DH + 128, DH);  // drop zero-ctx cols
  bias_sum_kernel<<<EW(G4), 256, 0, stream>>>(bias0, bih0, bhh0, G4);
  bias_sum_kernel<<<EW(G4), 256, 0, stream>>>(bias1, bih1, bhh1, G4);
  conv_pack_kernel<<<EW(5L * CMID * DSPEC_P), 256, 0, stream>>>(cw1p, cw1, CMID, CMID, DSPEC_P, DSPEC);
  conv_pack_kernel<<<EW(5L * CMID * CMID), 256, 0, stream>>>(cw2p, cw2, CMID, CMID, CMID, CMID);
  conv_pack_kernel<<<EW(5L * CMID * CMID), 256, 0, stream>>>(cw3p, cw3, CMID, CMID, CMID, CMID);
  conv_pack_kernel<<<EW(5L * CMID * CMID), 256, 0, stream>>>(cw4p, cw4, CMID, CMID, CMID, CMID);
  conv_pack_kernel<<<EW(5L * LIN_NP * CMID), 256, 0, stream>>>(cw5p, cw5, LIN_NP, DSPEC, CMID, CMID);

  // teacher-forcing input frames (bf16, padded)
  build_inbf_kernel<<<EW((long)MROWS * DSPEC_P), 256, 0, stream>>>(S_pad, inbf);

  // prenet: two WMMA GEMM+ReLU layers
  {
    int waves = MTILES * (DPRE / 16);
    gemm_relu_kernel<<<(waves + 7) / 8, 256, 0, stream>>>(inbf, DSPEC_P, preW1bf, pre_b1, pre1bf);
    gemm_relu_kernel<<<(waves + 7) / 8, 256, 0, stream>>>(pre1bf, DPRE, preW2bf, pre_b2, prebf);
  }

  // 2-layer LSTM recurrence, persistent grid, 2 global barriers/step,
  // in-register cell, LDS-shared A slabs
  lstm_kernel<<<NB_LSTM, 256, 0, stream>>>(prebf, Wih0bf, Whh0bf, bias0,
                                           Wih1bf, Whh1bf, bias1,
                                           h0buf, c0, h1buf, c1, outsbf, bar);

  // projection: S_pred -> d_out (f32) and xpadA (bf16, zero-margined)
  {
    int waves = MTILES * (LIN_NP / 16);
    linear_kernel<<<(waves + 7) / 8, 256, 0, stream>>>(outsbf, linWbf, lin_b, dout, xpadA);
  }

  // postnet: 5 conv layers as shifted WMMA GEMMs
  {
    int w512 = BATCH * LT_TILES * (CMID / 16);
    int w560 = BATCH * LT_TILES * (LIN_NP / 16);
    conv_kernel<<<(w512 + 7) / 8, 256, 0, stream>>>(xpadA, DSPEC_P, cw1p, CMID, CMID, cb1, xpadB, nullptr, 1);
    conv_kernel<<<(w512 + 7) / 8, 256, 0, stream>>>(xpadB, CMID, cw2p, CMID, CMID, cb2, xpadC, nullptr, 1);
    conv_kernel<<<(w512 + 7) / 8, 256, 0, stream>>>(xpadC, CMID, cw3p, CMID, CMID, cb3, xpadB, nullptr, 1);
    conv_kernel<<<(w512 + 7) / 8, 256, 0, stream>>>(xpadB, CMID, cw4p, CMID, CMID, cb4, xpadC, nullptr, 1);
    conv_kernel<<<(w560 + 7) / 8, 256, 0, stream>>>(xpadC, CMID, cw5p, LIN_NP, DSPEC, cb5, nullptr, dout, 0);
  }
}
